// AttnBlock_69836168233718
// MI455X (gfx1250) — matrix operand order repaired and checked by reading, not compiled, not run on hardware
//
#include <hip/hip_runtime.h>

typedef _Float16 half_t;
typedef __attribute__((ext_vector_type(16))) _Float16 v16h;
typedef __attribute__((ext_vector_type(8)))  _Float16 v8h;
typedef __attribute__((ext_vector_type(8)))  float    v8f;

#define BATCH 2
#define SEQ   2048
#define DIM   1024
#define NHEAD 16
#define NKV   4
#define HD    64
#define WIN   1024
#define RMS_EPS 1.1920929e-07f

// ---- gfx1250 async global->LDS path (ASYNCcnt), with safe fallback ----
#if defined(__has_builtin)
#  if __has_builtin(__builtin_amdgcn_global_load_async_to_lds_b128)
#    define HAS_ASYNC_LDS 1
#  endif
#endif
#ifndef HAS_ASYNC_LDS
#  define HAS_ASYNC_LDS 0
#endif

__device__ inline void stage_b128(const half_t* __restrict__ gsrc, half_t* lds_dst) {
#if HAS_ASYNC_LDS
  typedef int v4i __attribute__((vector_size(16)));
  typedef __attribute__((address_space(1))) v4i* gp_b128;   // global v4i*
  typedef __attribute__((address_space(3))) v4i* lp_b128;   // LDS v4i*
  __builtin_amdgcn_global_load_async_to_lds_b128(
      (gp_b128)(void*)gsrc, (lp_b128)(void*)lds_dst, 0, 0);
#else
  *(v8h*)lds_dst = *(const v8h*)gsrc;
#endif
}

// ---------- wave helpers (wave32) ----------
__device__ inline float wave_sum32(float v) {
  v += __shfl_xor(v, 16, 32); v += __shfl_xor(v, 8, 32);
  v += __shfl_xor(v, 4, 32);  v += __shfl_xor(v, 2, 32);
  v += __shfl_xor(v, 1, 32);
  return v;
}
__device__ inline float grp16_max(float v) {
  v = fmaxf(v, __shfl_xor(v, 1, 32)); v = fmaxf(v, __shfl_xor(v, 2, 32));
  v = fmaxf(v, __shfl_xor(v, 4, 32)); v = fmaxf(v, __shfl_xor(v, 8, 32));
  return v;
}
__device__ inline float grp16_sum(float v) {
  v += __shfl_xor(v, 1, 32); v += __shfl_xor(v, 2, 32);
  v += __shfl_xor(v, 4, 32); v += __shfl_xor(v, 8, 32);
  return v;
}

// ---------- WMMA operand loaders (contiguous k order, same for A and B) ----------
// A (16x32 f16): lane holds row lane%16; K = {0..15} (lanes<16) or {16..31}
__device__ inline v16h load_a16(const half_t* tile, int ld, int k0, int lane) {
  const half_t* p = tile + (size_t)(lane & 15) * ld + k0 + ((lane & 16) ? 16 : 0);
  v8h lo = *(const v8h*)p;
  v8h hi = *(const v8h*)(p + 8);
  v16h r;
#pragma unroll
  for (int i = 0; i < 8; ++i) { r[i] = lo[i]; r[i + 8] = hi[i]; }
  return r;
}
// B (32x16 f16): lane holds col lane%16; contiguous K run of 16 (lower/upper half)
__device__ inline v16h load_b16(const half_t* tile, int ld, int k0, int lane) {
  const half_t* p = tile + (size_t)(lane & 15) * ld + k0 + ((lane & 16) ? 16 : 0);
  return *(const v16h*)p;
}
__device__ inline v8f wmma_f16(v16h a, v16h b, v8f c) {
  return __builtin_amdgcn_wmma_f32_16x16x32_f16(false, a, false, b, (short)0, c, false, false);
}

// ---------- kernel 1: f32 -> f16 convert ----------
__global__ void k_convert(const float* __restrict__ in, half_t* __restrict__ out, int n) {
  int i = blockIdx.x * blockDim.x + threadIdx.x;
  if (i < n) out[i] = (half_t)in[i];
}

// ---------- kernel 2: RMSNorm over D, output f16 ----------
__global__ void k_rmsnorm(const float* __restrict__ x, half_t* __restrict__ h16) {
  const int row  = blockIdx.x;                    // B*L rows
  const int lane = threadIdx.x & 31, wave = threadIdx.x >> 5;
  const float* xr = x + (size_t)row * DIM;
  float ss = 0.f;
  for (int i = threadIdx.x; i < DIM; i += 256) { float v = xr[i]; ss += v * v; }
  ss = wave_sum32(ss);
  __shared__ float red[8];
  __shared__ float rinv;
  if (lane == 0) red[wave] = ss;
  __syncthreads();
  if (wave == 0) {
    float t = (lane < 8) ? red[lane] : 0.f;
    t += __shfl_xor(t, 4, 32); t += __shfl_xor(t, 2, 32); t += __shfl_xor(t, 1, 32);
    if (lane == 0) rinv = rsqrtf(t / (float)DIM + RMS_EPS);
  }
  __syncthreads();
  float r = rinv;
  for (int i = threadIdx.x; i < DIM; i += 256)
    h16[(size_t)row * DIM + i] = (half_t)(xr[i] * r);
}

// ---------- kernel 3: segment-id inclusive scan per batch row ----------
__global__ void k_segscan(const int* __restrict__ mask, int* __restrict__ seg) {
  const int b = blockIdx.x, t = threadIdx.x;      // 1024 threads, 2 elems each
  __shared__ int sc[1024];
  int m0 = mask[b * SEQ + 2 * t], m1 = mask[b * SEQ + 2 * t + 1];
  int ps = m0 + m1;
  sc[t] = ps;
  __syncthreads();
  for (int off = 1; off < 1024; off <<= 1) {
    int v = (t >= off) ? sc[t - off] : 0;
    __syncthreads();
    sc[t] += v;
    __syncthreads();
  }
  int excl = sc[t] - ps;
  seg[b * SEQ + 2 * t]     = excl + m0;
  seg[b * SEQ + 2 * t + 1] = excl + m0 + m1;
}

// ---------- kernel 4/7: WMMA GEMM  C[M,N] = A[M,K] * W[N,K]^T (+resid) ----------
// 8 waves/block. Block tile: 128 rows x 128 cols. Weight tile (128 cols x 32 K)
// is staged once per block into LDS with async global->LDS DMA, double-buffered.
// Each wave: 16 rows x 128 cols = 8 accumulators; one A fetch feeds 8 WMMAs.
__global__ void k_gemm_wmma(const half_t* __restrict__ A, const half_t* __restrict__ W,
                            float* __restrict__ C, const float* __restrict__ resid,
                            int M, int N, int K) {
  __shared__ half_t Bs[2][128 * 32];              // 2 x 8 KB
  const int tid = threadIdx.x;
  const int lane = tid & 31, wave = tid >> 5;
  const int row0 = (blockIdx.x * 8 + wave) * 16;
  const int col0 = blockIdx.y * 128;
  const half_t* Atile = A + (size_t)row0 * K;
  v8f acc[8] = {};

  // stage 512 x 16B per k-step: thread handles 2 chunks
  auto issue_stage = [&](int buf, int k0) {
#pragma unroll
    for (int j = 0; j < 2; ++j) {
      int idx = tid + j * 256;                    // 0..511
      int r = idx >> 2, q = idx & 3;              // W row (out col), 16B quarter
      stage_b128(W + (size_t)(col0 + r) * K + k0 + q * 8, &Bs[buf][r * 32 + q * 8]);
    }
  };

  issue_stage(0, 0);
  for (int k0 = 0; k0 < K; k0 += 32) {
    const int cur = (k0 >> 5) & 1;
    const bool more = (k0 + 32) < K;
    if (more) issue_stage(cur ^ 1, k0 + 32);
#if HAS_ASYNC_LDS
    if (more) asm volatile("s_wait_asynccnt 0x2" ::: "memory");
    else      asm volatile("s_wait_asynccnt 0x0" ::: "memory");
#endif
    __syncthreads();                              // tile visible to all waves
    v16h a = load_a16(Atile, K, k0, lane);
#pragma unroll
    for (int t = 0; t < 8; ++t) {
      v16h b = load_b16(&Bs[cur][t * 16 * 32], 32, 0, lane);
      acc[t] = wmma_f16(a, b, acc[t]);
    }
    __syncthreads();                              // all reads done before rewrite
  }

  const int n_l = lane & 15, mb = (lane & 16) ? 8 : 0;
#pragma unroll
  for (int t = 0; t < 8; ++t)
#pragma unroll
    for (int i = 0; i < 8; ++i) {
      size_t idx = (size_t)(row0 + mb + i) * N + col0 + t * 16 + n_l;
      float v = acc[t][i];
      if (resid) v += resid[idx];
      C[idx] = v;
    }
}

// ---------- kernel 5: per-head RMSNorm + RoPE, layout for attention ----------
// one wave per (b,l,head-instance); lane t owns rope pair (t, t+32)
__global__ void k_qknorm_rope(const float* __restrict__ qkv, half_t* __restrict__ q16,
                              half_t* __restrict__ k16, half_t* __restrict__ vT16) {
  const int t = threadIdx.x;                      // 0..31
  int gid = blockIdx.x;
  const int hh = gid % 24;
  const int l  = (gid / 24) % SEQ;
  const int b  = gid / (24 * SEQ);
  const float* row = qkv + (size_t)(b * SEQ + l) * 1536;
  int base = (hh < 16) ? hh * 64 : (hh < 20) ? 1024 + (hh - 16) * 64 : 1280 + (hh - 20) * 64;
  float xa = row[base + t], xb = row[base + t + 32];
  if (hh < 20) {
    float ss = wave_sum32(xa * xa + xb * xb);
    float r = rsqrtf(ss / 64.f + RMS_EPS);
    xa *= r; xb *= r;
    float inv = __expf(-(float)t * (9.210340372f / 32.f));  // 10000^(-t/32)
    float f = (float)l * inv;
    float cf = cosf(f), sf = sinf(f);
    float o1 = xa * cf - xb * sf;
    float o2 = xa * sf + xb * cf;
    if (hh < 16) {
      half_t* Q = q16 + (((size_t)(b * NHEAD + hh)) * SEQ + l) * HD;
      Q[t] = (half_t)o1; Q[t + 32] = (half_t)o2;
    } else {
      half_t* Kp = k16 + (((size_t)(b * NKV + (hh - 16))) * SEQ + l) * HD;
      Kp[t] = (half_t)o1; Kp[t + 32] = (half_t)o2;
    }
  } else {
    half_t* V = vT16 + ((size_t)(b * NKV + (hh - 20))) * HD * SEQ;  // [dim][L]
    V[(size_t)t * SEQ + l]        = (half_t)xa;
    V[(size_t)(t + 32) * SEQ + l] = (half_t)xb;
  }
}

// ---------- kernel 6: flash attention (sliding window + doc mask) ----------
// one wave per (b,h,16-row q tile); 4 waves per block
__global__ void k_attn(const half_t* __restrict__ q16, const half_t* __restrict__ k16,
                       const half_t* __restrict__ vT16, const int* __restrict__ seg,
                       half_t* __restrict__ y16) {
  const int lane = threadIdx.x & 31, wave = threadIdx.x >> 5;
  const int gid = blockIdx.x * 4 + wave;
  const int qt = gid & (SEQ / 16 - 1);
  const int h  = (gid >> 7) & (NHEAD - 1);
  const int b  = gid >> 11;
  const int kvh = h >> 2;                         // repeat(k,4): head h -> kv h/4
  const int q0 = qt * 16;

  const half_t* Q  = q16 + (((size_t)(b * NHEAD + h)) * SEQ + q0) * HD;
  const half_t* Kp = k16 + ((size_t)(b * NKV + kvh)) * SEQ * HD;
  const half_t* Vt = vT16 + ((size_t)(b * NKV + kvh)) * HD * SEQ;
  const int* segb = seg + b * SEQ;

  __shared__ half_t Pbuf[4][16 * 32];
  half_t* P = Pbuf[wave];

  const v16h qa0 = load_a16(Q, HD, 0, lane);      // dims 0..31
  const v16h qa1 = load_a16(Q, HD, 32, lane);     // dims 32..63

  const int n_l = lane & 15, mb = (lane & 16) ? 8 : 0;
  int   segq[8];
  float Mr[8], Lr[8];
#pragma unroll
  for (int i = 0; i < 8; ++i) { segq[i] = segb[q0 + mb + i]; Mr[i] = -1e30f; Lr[i] = 0.f; }
  v8f O[4] = {};

  int kstart = q0 - (WIN - 1);
  if (kstart < 0) kstart = 0;
  kstart &= ~31;

  for (int kb = kstart; kb < q0 + 16; kb += 32) {
    // ---- scores: two 16-key tiles, K-dim = 64 (2 wmma each) ----
    v8f s0 = {}, s1 = {};
    s0 = wmma_f16(qa0, load_b16(Kp + (size_t)kb * HD, HD, 0, lane), s0);
    s0 = wmma_f16(qa1, load_b16(Kp + (size_t)kb * HD, HD, 32, lane), s0);
    s1 = wmma_f16(qa0, load_b16(Kp + (size_t)(kb + 16) * HD, HD, 0, lane), s1);
    s1 = wmma_f16(qa1, load_b16(Kp + (size_t)(kb + 16) * HD, HD, 32, lane), s1);

    const int kpos0 = kb + n_l, kpos1 = kb + 16 + n_l;
    const int sk0 = segb[kpos0], sk1 = segb[kpos1];

    float corr[8];
#pragma unroll
    for (int i = 0; i < 8; ++i) {
      int qpos = q0 + mb + i;
      int d0 = qpos - kpos0, d1 = qpos - kpos1;
      float v0 = ((unsigned)d0 < (unsigned)WIN && segq[i] == sk0) ? s0[i] * 0.125f : -1e30f;
      float v1 = ((unsigned)d1 < (unsigned)WIN && segq[i] == sk1) ? s1[i] * 0.125f : -1e30f;
      float mnew = fmaxf(Mr[i], grp16_max(fmaxf(v0, v1)));
      float c = __expf(Mr[i] - mnew);
      float p0 = (v0 > -1e29f) ? __expf(v0 - mnew) : 0.f;
      float p1 = (v1 > -1e29f) ? __expf(v1 - mnew) : 0.f;
      Lr[i] = Lr[i] * c + grp16_sum(p0 + p1);
      Mr[i] = mnew; corr[i] = c;
      P[(mb + i) * 32 + n_l]      = (half_t)p0;   // stage probs for A-operand
      P[(mb + i) * 32 + 16 + n_l] = (half_t)p1;
    }
#pragma unroll
    for (int t = 0; t < 4; ++t)
#pragma unroll
      for (int i = 0; i < 8; ++i) O[t][i] *= corr[i];

    asm volatile("s_wait_dscnt 0" ::: "memory");  // P stores visible wave-wide

    const v16h pa = load_a16(P, 32, 0, lane);     // P as A (16 rows x 32 keys)
#pragma unroll
    for (int t = 0; t < 4; ++t) {
      v16h bv = load_b16(Vt + (size_t)(t * 16) * SEQ, SEQ, kb, lane);  // V^T contiguous keys
      O[t] = wmma_f16(pa, bv, O[t]);
    }
  }

#pragma unroll
  for (int t = 0; t < 4; ++t)
#pragma unroll
    for (int i = 0; i < 8; ++i) {
      int m = q0 + mb + i;
      y16[((size_t)(b * SEQ + m)) * DIM + h * HD + t * 16 + n_l] = (half_t)(O[t][i] / Lr[i]);
    }
}

// ---------- host ----------
extern "C" void kernel_launch(void* const* d_in, const int* in_sizes, int n_in,
                              void* d_out, int out_size, void* d_ws, size_t ws_size,
                              hipStream_t stream) {
  (void)in_sizes; (void)n_in; (void)out_size; (void)ws_size;
  const float* x     = (const float*)d_in[0];
  const float* w_qkv = (const float*)d_in[1];
  const float* w_out = (const float*)d_in[2];
  const int*   reset = (const int*)d_in[3];
  float* out = (float*)d_out;

  char* ws = (char*)d_ws;
  size_t off = 0;
  auto alloc = [&](size_t bytes) {
    void* p = ws + off;
    off += (bytes + 255) & ~(size_t)255;
    return p;
  };
  const size_t ML = (size_t)BATCH * SEQ;          // 4096 rows
  half_t* h16  = (half_t*)alloc(ML * DIM * 2);
  half_t* wq16 = (half_t*)alloc((size_t)1536 * DIM * 2);
  half_t* wo16 = (half_t*)alloc((size_t)DIM * DIM * 2);
  float*  qkvf = (float*)alloc(ML * 1536 * 4);
  half_t* q16  = (half_t*)alloc(ML * DIM * 2);
  half_t* k16  = (half_t*)alloc((size_t)BATCH * NKV * SEQ * HD * 2);
  half_t* vT16 = (half_t*)alloc((size_t)BATCH * NKV * SEQ * HD * 2);
  half_t* y16  = (half_t*)alloc(ML * DIM * 2);
  int*    seg  = (int*)alloc((size_t)BATCH * SEQ * 4);

  // 1) weights -> f16
  k_convert<<<(1536 * DIM + 255) / 256, 256, 0, stream>>>(w_qkv, wq16, 1536 * DIM);
  k_convert<<<(DIM * DIM + 255) / 256, 256, 0, stream>>>(w_out, wo16, DIM * DIM);
  // 2) rmsnorm(x) -> h16
  k_rmsnorm<<<(int)ML, 256, 0, stream>>>(x, h16);
  // 3) segment ids
  k_segscan<<<BATCH, 1024, 0, stream>>>(reset, seg);
  // 4) qkv = h @ w_qkv^T  (f32 out for norm/rope precision)
  k_gemm_wmma<<<dim3(ML / 128, 1536 / 128), 256, 0, stream>>>(h16, wq16, qkvf, nullptr,
                                                              (int)ML, 1536, DIM);
  // 5) per-head rmsnorm + rope -> q16/k16/vT16
  k_qknorm_rope<<<BATCH * SEQ * 24, 32, 0, stream>>>(qkvf, q16, k16, vT16);
  // 6) windowed doc-masked flash attention -> y16
  k_attn<<<(BATCH * NHEAD * (SEQ / 16)) / 4, 128, 0, stream>>>(q16, k16, vT16, seg, y16);
  // 7) out = x + y @ w_out^T
  k_gemm_wmma<<<dim3(ML / 128, DIM / 128), 256, 0, stream>>>(y16, wo16, out, x,
                                                             (int)ML, DIM, DIM);
}
